// HORSTCell_51582557225382
// MI455X (gfx1250) — compile-verified, matrix operand order checked by reading, not run on hardware
//
#include <hip/hip_runtime.h>
#include <hip/hip_bf16.h>
#include <math.h>

// ---------------------------------------------------------------------------
// Problem constants (match reference): B=16, Cin=64, C=64, S=8, H=W=64, R=16
// ---------------------------------------------------------------------------
#define HH    64
#define WWID  64
#define HWSZ  4096                  // H*W
#define CC    64
#define CHWSZ (CC * HWSZ)           // 262144
#define BB    16
#define SS    8

typedef __attribute__((ext_vector_type(16))) _Float16 v16h;
typedef __attribute__((ext_vector_type(8)))  float    v8f;
typedef __attribute__((ext_vector_type(4)))  float    f32x4;

// ---------------------------------------------------------------------------
// Weight pre-transform: OIHW fp32 -> [khw][co][ci] fp16 (contiguous in ci so
// LDS staging is coalesced b128 and A-fragments are 2x ds_load_b128).
// ---------------------------------------------------------------------------
__global__ void wprep_kernel(const float* __restrict__ wgt,  // [64,CinT,3,3]
                             _Float16* __restrict__ o,       // [9][64][CinT]
                             int CinT)
{
    const int total = 64 * CinT * 9;
    int i = blockIdx.x * blockDim.x + threadIdx.x;
    if (i >= total) return;
    const int ci  = i & (CinT - 1);          // CinT is 64 or 128 (pow2)
    const int t   = i / CinT;
    const int co  = t & 63;
    const int khw = t >> 6;
    o[i] = (_Float16)wgt[(co * CinT + ci) * 9 + khw];
}

// ---------------------------------------------------------------------------
// Implicit-GEMM 3x3 conv (pad=1) with WMMA f16->f32, compile-time channel
// counts so every LDS / global fragment load uses immediate offsets.
// Loop order: tap (kh,kw) outer x 32-channel K-chunks inner (fully unrolled).
//   - per-tap weights staged to LDS, A-frag = 2x ds_load_b128 off one base
//   - B-frag = 16 global dword loads at constant channel-stride offsets
//   - y-bounds wave-uniform, x-bounds one predicate per fragment
// M = Co = 64 (4 tiles), N = 128 px/WG (4 waves x 2 tiles of 16).
// Channel concat: chunk base < CIN_A -> srcA else srcB (compile-time select).
// Also accumulates GroupNorm sum/sumsq.
// ---------------------------------------------------------------------------
template<int CIN_T, int CIN_A>
__global__ __launch_bounds__(128)
void conv3x3_wmma_kernel(const float* __restrict__ srcA,
                         const float* __restrict__ srcB,
                         const _Float16* __restrict__ wgt16, // [9][64][CIN_T]
                         float* __restrict__ out,            // [Nimg,64,H,W]
                         float* __restrict__ stats)          // [Nimg][2]
{
    constexpr int NCIC  = CIN_T >> 5;         // 32-channel K-chunks
    constexpr int NF4   = CIN_T * 8;          // float4s per 64xCIN_T f16 slice
    constexpr int CIN_B = CIN_T - CIN_A;

    __shared__ _Float16 wl[64 * CIN_T];       // one tap slice (8/16 KB)
    __shared__ float rs1[128], rs2[128];

    const int tid  = threadIdx.x;
    const int lane = tid & 31;
    const int wave = tid >> 5;
    const int img  = blockIdx.x >> 5;         // 4096 px / 128 px-per-WG
    const int pix0 = (blockIdx.x & 31) * 128;

    const int n15   = lane & 15;
    const int kbase = (lane < 16) ? 0 : 8;    // 16-bit fragment K striping

    // N-tile geometry: tiles of 16 consecutive pixels never cross a row.
    int pix_nt[2], py[2], px[2];
#pragma unroll
    for (int nt = 0; nt < 2; ++nt) {
        const int pbase = pix0 + wave * 32 + nt * 16;
        pix_nt[nt] = pbase + n15;
        py[nt] = pbase >> 6;                  // uniform within the wave
        px[nt] = (pbase & 63) + n15;          // per-lane column
    }

    v8f acc[4][2];
#pragma unroll
    for (int t = 0; t < 4; ++t)
#pragma unroll
        for (int nt = 0; nt < 2; ++nt)
            acc[t][nt] = (v8f){0.f,0.f,0.f,0.f,0.f,0.f,0.f,0.f};

    // per-lane source bases (include K-striping half offset)
    const float* pA = srcA + (size_t)img * CIN_A * HWSZ + (size_t)kbase * HWSZ;
    const float* pB = (CIN_B > 0)
        ? (srcB + (size_t)img * CIN_B * HWSZ + (size_t)kbase * HWSZ)
        : pA;

    // single LDS base for all A-fragment loads (offsets are immediates)
    const _Float16* wp0 = wl + n15 * CIN_T + kbase;

    for (int khw = 0; khw < 9; ++khw) {
        const int kh = khw / 3;               // scalar-unit math only
        const int kw = khw - kh * 3;

        // --- stage this tap's 64xCIN_T f16 weight slice into LDS (coalesced)
        __syncthreads();
        {
            const f32x4* src = (const f32x4*)(wgt16 + (size_t)khw * 64 * CIN_T);
            f32x4* dst = (f32x4*)wl;
#pragma unroll
            for (int j = 0; j < NF4 / 128; ++j)
                dst[j * 128 + tid] = src[j * 128 + tid];
            if (khw < 8)
                __builtin_prefetch(src + NF4, 0, 1);   // next tap slice
        }
        __syncthreads();

        // --- per-N-tile tap geometry (hoisted out of the K-chunk loop)
        int xoff[2];
        bool yok[2], xok[2];
#pragma unroll
        for (int nt = 0; nt < 2; ++nt) {
            const int y2 = py[nt] + kh - 1;
            yok[nt] = (unsigned)y2 < (unsigned)HH;      // wave-uniform
            const int x2 = px[nt] + kw - 1;
            xok[nt] = (unsigned)x2 < (unsigned)WWID;    // per-lane
            xoff[nt] = y2 * WWID + x2;
        }

#pragma unroll
        for (int cic = 0; cic < NCIC; ++cic) {
            // A fragments: 4 Co tiles, 2x b128 LDS loads each, one base addr
            v16h a[4];
#pragma unroll
            for (int t = 0; t < 4; ++t) {
                const _Float16* wp = wp0 + t * 16 * CIN_T + cic * 32;
                union { v16h v; f32x4 f[2]; } u;
                u.f[0] = *(const f32x4*)wp;        // K = kbase+0..7
                u.f[1] = *(const f32x4*)(wp + 16); // K = 16+kbase+0..7
                a[t] = u.v;
            }

            // input channel-chunk base (concat select is compile-time)
            const float* sbase = (cic * 32 < CIN_A)
                ? (pA + (size_t)(cic * 32) * HWSZ)
                : (pB + (size_t)(cic * 32 - CIN_A) * HWSZ);

#pragma unroll
            for (int nt = 0; nt < 2; ++nt) {
                if (!yok[nt]) continue;                 // wave-uniform skip
                v16h b;
                const float* q = sbase + xoff[nt];
                if (xok[nt]) {
#pragma unroll
                    for (int e = 0; e < 8; ++e) {
                        b[e]     = (_Float16)q[(size_t)e * HWSZ];
                        b[e + 8] = (_Float16)q[(size_t)(16 + e) * HWSZ];
                    }
                } else {
#pragma unroll
                    for (int e = 0; e < 16; ++e) b[e] = (_Float16)0.f;
                }
#pragma unroll
                for (int t = 0; t < 4; ++t) {
                    acc[t][nt] = __builtin_amdgcn_wmma_f32_16x16x32_f16(
                        false, a[t], false, b,
                        (short)0, acc[t][nt], false, false);
                }
            }
        }
    }

    // --- store D tiles + accumulate GroupNorm partial statistics
    float s1 = 0.f, s2 = 0.f;
    const size_t obase = (size_t)img * CHWSZ;
    const int mhalf = (lane < 16) ? 0 : 8;
#pragma unroll
    for (int t = 0; t < 4; ++t) {
#pragma unroll
        for (int nt = 0; nt < 2; ++nt) {
#pragma unroll
            for (int r = 0; r < 8; ++r) {
                const int m = t * 16 + r + mhalf;
                const float v = acc[t][nt][r];
                out[obase + (size_t)m * HWSZ + pix_nt[nt]] = v;
                s1 += v; s2 += v * v;
            }
        }
    }
    rs1[tid] = s1; rs2[tid] = s2;
    __syncthreads();
    for (int off = 64; off > 0; off >>= 1) {
        if (tid < off) { rs1[tid] += rs1[tid + off]; rs2[tid] += rs2[tid + off]; }
        __syncthreads();
    }
    if (tid == 0) {
        atomicAdd(&stats[img * 2 + 0], rs1[0]);
        atomicAdd(&stats[img * 2 + 1], rs2[0]);
    }
}

// ---------------------------------------------------------------------------
// GroupNorm(1, C) apply (in-place), optional SiLU.
// ---------------------------------------------------------------------------
__global__ void gn_apply_kernel(float* __restrict__ x,
                                const float* __restrict__ stats,
                                const float* __restrict__ g,
                                const float* __restrict__ b,
                                int silu, int total)
{
    int i = blockIdx.x * blockDim.x + threadIdx.x;
    if (i >= total) return;
    const int img = i / CHWSZ;
    const int c   = (i - img * CHWSZ) >> 12;
    const float cnt = (float)CHWSZ;
    const float mu  = stats[img * 2 + 0] / cnt;
    const float var = stats[img * 2 + 1] / cnt - mu * mu;
    const float inv = rsqrtf(var + 1e-5f);
    float v = (x[i] - mu) * inv * g[c] + b[c];
    if (silu) v = v / (1.f + __expf(-v));
    x[i] = v;
}

// ---------------------------------------------------------------------------
// Spatial filter: channel mean/max maps, then tiny 2->1 3x3 conv + sigmoid.
// ---------------------------------------------------------------------------
__global__ void meanmax_kernel(const float* __restrict__ x,
                               float* __restrict__ mm, int Nimg)
{
    int i = blockIdx.x * blockDim.x + threadIdx.x;
    if (i >= Nimg * HWSZ) return;
    const int img = i >> 12, pix = i & 4095;
    const float* p = x + (size_t)img * CHWSZ + pix;
    float s = 0.f, mx = -1e30f;
#pragma unroll 8
    for (int c = 0; c < CC; ++c) { float v = p[(size_t)c * HWSZ]; s += v; mx = fmaxf(mx, v); }
    mm[(size_t)img * 2 * HWSZ + pix]        = s * (1.f / CC);
    mm[(size_t)img * 2 * HWSZ + HWSZ + pix] = mx;
}

__global__ void sfconv_kernel(const float* __restrict__ mm,
                              const float* __restrict__ w,
                              const float* __restrict__ bias,
                              float* __restrict__ f, int Nimg)
{
    int i = blockIdx.x * blockDim.x + threadIdx.x;
    if (i >= Nimg * HWSZ) return;
    const int img = i >> 12, pix = i & 4095;
    const int y = pix >> 6, x = pix & 63;
    float acc = bias[0];
#pragma unroll
    for (int c = 0; c < 2; ++c)
#pragma unroll
        for (int kh = 0; kh < 3; ++kh)
#pragma unroll
            for (int kw = 0; kw < 3; ++kw) {
                const int y2 = y + kh - 1, x2 = x + kw - 1;
                if ((unsigned)y2 < (unsigned)HH && (unsigned)x2 < (unsigned)WWID)
                    acc += mm[(size_t)img * 2 * HWSZ + (size_t)c * HWSZ + y2 * WWID + x2]
                         * w[c * 9 + kh * 3 + kw];
            }
    f[i] = 1.f / (1.f + __expf(-acc));
}

// f_KQ[s,b,c] = mean_pix f_K[sb,pix] * Qp[b,c,pix]
__global__ void fkq_kernel(const float* __restrict__ fK,
                           const float* __restrict__ Qp,
                           float* __restrict__ fkq)
{
    __shared__ float red[256];
    const int sbc = blockIdx.x;
    const int sb = sbc >> 6, c = sbc & 63;
    const int b = sb & 15;
    const float* fk = fK + (size_t)sb * HWSZ;
    const float* q  = Qp + ((size_t)b * CC + c) * HWSZ;
    float s = 0.f;
    for (int p = threadIdx.x; p < HWSZ; p += 256) s += fk[p] * q[p];
    red[threadIdx.x] = s; __syncthreads();
    for (int off = 128; off > 0; off >>= 1) {
        if ((int)threadIdx.x < off) red[threadIdx.x] += red[threadIdx.x + off];
        __syncthreads();
    }
    if (threadIdx.x == 0) fkq[sbc] = red[0] * (1.f / HWSZ);
}

// gate[s,b,pix] = sigmoid( sum_c fkq[s,b,c] * Kp[sb,c,pix] )
__global__ void gate_kernel(const float* __restrict__ fkq,
                            const float* __restrict__ Kp,
                            float* __restrict__ gate)
{
    __shared__ float fc[CC];
    const int sb  = blockIdx.x >> 4;
    const int pix = (blockIdx.x & 15) * 256 + threadIdx.x;
    if (threadIdx.x < CC) fc[threadIdx.x] = fkq[sb * CC + threadIdx.x];
    __syncthreads();
    const float* kp = Kp + (size_t)sb * CHWSZ + pix;
    float s = 0.f;
#pragma unroll 8
    for (int c = 0; c < CC; ++c) s += fc[c] * kp[(size_t)c * HWSZ];
    gate[(size_t)sb * HWSZ + pix] = 1.f / (1.f + __expf(-s));
}

// logits[s,b] = d^-0.5 * sum_{c,pix} (Qp*fQ)[b,c,pix] * (Kp*fK)[sb,c,pix]
__global__ void logits_kernel(const float* __restrict__ Qp,
                              const float* __restrict__ fQ,
                              const float* __restrict__ Kp,
                              const float* __restrict__ fK,
                              float* __restrict__ logits)
{
    __shared__ float red[256];
    const int sb = blockIdx.x, b = sb & 15;
    float s = 0.f;
    for (int i = threadIdx.x; i < CHWSZ; i += 256) {
        const int pix = i & 4095;
        const float qf = Qp[(size_t)b  * CHWSZ + i] * fQ[b  * HWSZ + pix];
        const float kf = Kp[(size_t)sb * CHWSZ + i] * fK[sb * HWSZ + pix];
        s += qf * kf;
    }
    red[threadIdx.x] = s; __syncthreads();
    for (int off = 128; off > 0; off >>= 1) {
        if ((int)threadIdx.x < off) red[threadIdx.x] += red[threadIdx.x + off];
        __syncthreads();
    }
    if (threadIdx.x == 0) logits[sb] = red[0] * (1.f / 512.f);  // 262144^-0.5
}

__global__ void softmax_kernel(const float* __restrict__ logits,
                               float* __restrict__ T)
{
    const int b = threadIdx.x;
    if (b >= BB) return;
    float mx = -1e30f;
#pragma unroll
    for (int s = 0; s < SS; ++s) mx = fmaxf(mx, logits[s * BB + b]);
    float e[SS], sum = 0.f;
#pragma unroll
    for (int s = 0; s < SS; ++s) { e[s] = __expf(logits[s * BB + b] - mx); sum += e[s]; }
#pragma unroll
    for (int s = 0; s < SS; ++s) T[s * BB + b] = e[s] / sum;
}

// att = sum_s T[s,b] * Vp[sb,c,pix] * gate[s,b,pix]
// new_state = mask[b] ? att : 0 ; aht = att + ht
__global__ void att_kernel(const float* __restrict__ T,
                           const float* __restrict__ Vp,
                           const float* __restrict__ gate,
                           const float* __restrict__ ht,
                           const unsigned char* __restrict__ mask,
                           float* __restrict__ new_state,
                           float* __restrict__ aht)
{
    const int i = blockIdx.x * 256 + threadIdx.x;        // b*CHW + c*HW + pix
    const int b   = i >> 18;                             // CHW = 2^18
    const int pix = i & 4095;
    const int chw = i & (CHWSZ - 1);
    float a = 0.f;
#pragma unroll
    for (int s = 0; s < SS; ++s) {
        const int sb = s * BB + b;
        a += T[sb] * Vp[(size_t)sb * CHWSZ + chw] * gate[(size_t)sb * HWSZ + pix];
    }
    new_state[i] = mask[b] ? a : 0.f;
    aht[i] = a + ht[i];
}

__global__ void pool_kernel(const float* __restrict__ h, float* __restrict__ pooled)
{
    __shared__ float red[256];
    const float* p = h + (size_t)blockIdx.x * HWSZ;
    float s = 0.f;
    for (int i = threadIdx.x; i < HWSZ; i += 256) s += p[i];
    red[threadIdx.x] = s; __syncthreads();
    for (int off = 128; off > 0; off >>= 1) {
        if ((int)threadIdx.x < off) red[threadIdx.x] += red[threadIdx.x + off];
        __syncthreads();
    }
    if (threadIdx.x == 0) pooled[blockIdx.x] = red[0] * (1.f / HWSZ);
}

__global__ void se_kernel(const float* __restrict__ pooled,
                          const float* __restrict__ w1, const float* __restrict__ b1,
                          const float* __restrict__ w2, const float* __restrict__ b2,
                          float* __restrict__ se)
{
    __shared__ float pc[CC], z[16];
    const int b = blockIdx.x, t = threadIdx.x;
    pc[t] = pooled[b * CC + t];
    __syncthreads();
    if (t < 16) {
        float s = b1[t];
#pragma unroll 8
        for (int c = 0; c < CC; ++c) s += w1[t * CC + c] * pc[c];
        z[t] = fmaxf(s, 0.f);
    }
    __syncthreads();
    float s = b2[t];
#pragma unroll
    for (int r = 0; r < 16; ++r) s += w2[t * 16 + r] * z[r];
    se[b * CC + t] = 1.f / (1.f + __expf(-s));
}

__global__ void final_kernel(const float* __restrict__ h,
                             const float* __restrict__ se,
                             float* __restrict__ outp)
{
    const int i = blockIdx.x * 256 + threadIdx.x;
    const int bc = i >> 12;
    const float v = h[i];
    outp[i] = v * (1.f + se[bc]);
}

// ---------------------------------------------------------------------------
// Host orchestration
// ---------------------------------------------------------------------------
extern "C" void kernel_launch(void* const* d_in, const int* in_sizes, int n_in,
                              void* d_out, int out_size, void* d_ws, size_t ws_size,
                              hipStream_t stream)
{
    (void)in_sizes; (void)n_in; (void)out_size; (void)ws_size;

    const float* inputs   = (const float*)d_in[0];
    const float* hiddenq  = (const float*)d_in[1];
    const float* outputq  = (const float*)d_in[2];
    const unsigned char* mask = (const unsigned char*)d_in[3];
    const float* enc_w = (const float*)d_in[4];
    const float* enc_g = (const float*)d_in[5];
    const float* enc_b = (const float*)d_in[6];
    const float* q_w   = (const float*)d_in[7];
    const float* q_g   = (const float*)d_in[8];
    const float* q_b   = (const float*)d_in[9];
    const float* k_w   = (const float*)d_in[10];
    const float* k_g   = (const float*)d_in[11];
    const float* k_b   = (const float*)d_in[12];
    const float* v_w   = (const float*)d_in[13];
    const float* v_g   = (const float*)d_in[14];
    const float* v_b   = (const float*)d_in[15];
    const float* sfq_w = (const float*)d_in[16];
    const float* sfq_b = (const float*)d_in[17];
    const float* sfk_w = (const float*)d_in[18];
    const float* sfk_b = (const float*)d_in[19];
    const float* out_w = (const float*)d_in[20];
    const float* out_g = (const float*)d_in[21];
    const float* out_b = (const float*)d_in[22];
    const float* sse_w1 = (const float*)d_in[23];
    const float* sse_b1 = (const float*)d_in[24];
    const float* sse_w2 = (const float*)d_in[25];
    const float* sse_b2 = (const float*)d_in[26];

    const size_t BCHW  = (size_t)BB * CHWSZ;      // 4,194,304
    const size_t SBCHW = (size_t)SS * BCHW;       // 33,554,432

    float* outputs   = (float*)d_out;
    float* ht        = outputs + BCHW;            // returned as 2nd output
    float* new_state = ht + BCHW;                 // returned as 3rd output

    float* w = (float*)d_ws;
    float* Qp    = w;                w += BCHW;
    float* Kp    = w;                w += SBCHW;
    float* Vp    = w;                w += SBCHW;
    float* mm    = w;                w += (size_t)SS * BB * 2 * HWSZ;
    float* fQ    = w;                w += (size_t)BB * HWSZ;
    float* fK    = w;                w += (size_t)SS * BB * HWSZ;
    float* fkq   = w;                w += SS * BB * CC;
    float* gate  = w;                w += (size_t)SS * BB * HWSZ;
    float* logit = w;                w += SS * BB;
    float* Tm    = w;                w += SS * BB;
    float* aht   = w;                w += BCHW;
    float* hbuf  = w;                w += BCHW;
    float* pooled= w;                w += BB * CC;
    float* seb   = w;                w += BB * CC;
    float* stats = w;                w += 2 * SS * BB;
    _Float16* wgt16 = (_Float16*)w;  // 9*64*128 halfs max (147456 B)

    const dim3 blk128(128), blk256(256), blk64(64), blk32(32);
    const int gElemB  = (int)(BCHW / 256);        // 16384
    const int gElemSB = (int)(SBCHW / 256);       // 131072
    const int gW64    = (9 * 64 * 64  + 255) / 256;
    const int gW128   = (9 * 64 * 128 + 255) / 256;

    // ---- encoder: conv -> GN -> SiLU  => ht (in d_out)
    hipMemsetAsync(stats, 0, 2 * SS * BB * sizeof(float), stream);
    wprep_kernel<<<gW64, blk256, 0, stream>>>(enc_w, wgt16, 64);
    conv3x3_wmma_kernel<64, 64><<<BB * 32, blk128, 0, stream>>>(inputs, nullptr,
                                                                wgt16, ht, stats);
    gn_apply_kernel<<<gElemB, blk256, 0, stream>>>(ht, stats, enc_g, enc_b, 1, (int)BCHW);

    // ---- Qp = GN(conv(ht))
    hipMemsetAsync(stats, 0, 2 * SS * BB * sizeof(float), stream);
    wprep_kernel<<<gW64, blk256, 0, stream>>>(q_w, wgt16, 64);
    conv3x3_wmma_kernel<64, 64><<<BB * 32, blk128, 0, stream>>>(ht, nullptr,
                                                                wgt16, Qp, stats);
    gn_apply_kernel<<<gElemB, blk256, 0, stream>>>(Qp, stats, q_g, q_b, 0, (int)BCHW);

    // ---- Kp = GN(conv(concat(hidden, output)))
    hipMemsetAsync(stats, 0, 2 * SS * BB * sizeof(float), stream);
    wprep_kernel<<<gW128, blk256, 0, stream>>>(k_w, wgt16, 128);
    conv3x3_wmma_kernel<128, 64><<<SS * BB * 32, blk128, 0, stream>>>(hiddenq, outputq,
                                                                      wgt16, Kp, stats);
    gn_apply_kernel<<<gElemSB, blk256, 0, stream>>>(Kp, stats, k_g, k_b, 0, (int)SBCHW);

    // ---- Vp = GN(conv(concat(hidden, output)))
    hipMemsetAsync(stats, 0, 2 * SS * BB * sizeof(float), stream);
    wprep_kernel<<<gW128, blk256, 0, stream>>>(v_w, wgt16, 128);
    conv3x3_wmma_kernel<128, 64><<<SS * BB * 32, blk128, 0, stream>>>(hiddenq, outputq,
                                                                      wgt16, Vp, stats);
    gn_apply_kernel<<<gElemSB, blk256, 0, stream>>>(Vp, stats, v_g, v_b, 0, (int)SBCHW);

    // ---- spatial filters
    meanmax_kernel<<<BB * HWSZ / 256, blk256, 0, stream>>>(Qp, mm, BB);
    sfconv_kernel<<<BB * HWSZ / 256, blk256, 0, stream>>>(mm, sfq_w, sfq_b, fQ, BB);
    meanmax_kernel<<<SS * BB * HWSZ / 256, blk256, 0, stream>>>(Kp, mm, SS * BB);
    sfconv_kernel<<<SS * BB * HWSZ / 256, blk256, 0, stream>>>(mm, sfk_w, sfk_b, fK, SS * BB);

    // ---- gating + attention over queue axis
    fkq_kernel<<<SS * BB * CC, blk256, 0, stream>>>(fK, Qp, fkq);
    gate_kernel<<<SS * BB * 16, blk256, 0, stream>>>(fkq, Kp, gate);
    logits_kernel<<<SS * BB, blk256, 0, stream>>>(Qp, fQ, Kp, fK, logit);
    softmax_kernel<<<1, blk32, 0, stream>>>(logit, Tm);
    att_kernel<<<gElemB, blk256, 0, stream>>>(Tm, Vp, gate, ht, mask, new_state, aht);

    // ---- out conv: conv(concat(inputs, att+ht)) -> GN -> SiLU  => h
    hipMemsetAsync(stats, 0, 2 * SS * BB * sizeof(float), stream);
    wprep_kernel<<<gW128, blk256, 0, stream>>>(out_w, wgt16, 128);
    conv3x3_wmma_kernel<128, 64><<<BB * 32, blk128, 0, stream>>>(inputs, aht,
                                                                 wgt16, hbuf, stats);
    gn_apply_kernel<<<gElemB, blk256, 0, stream>>>(hbuf, stats, out_g, out_b, 1, (int)BCHW);

    // ---- squeeze-and-excitation + residual
    pool_kernel<<<BB * CC, blk256, 0, stream>>>(hbuf, pooled);
    se_kernel<<<BB, blk64, 0, stream>>>(pooled, sse_w1, sse_b1, sse_w2, sse_b2, seb);
    final_kernel<<<gElemB, blk256, 0, stream>>>(hbuf, seb, outputs);
}
